// RecNet_223338299583
// MI455X (gfx1250) — matrix operand order repaired and checked by reading, not compiled, not run on hardware
//
#include <hip/hip_runtime.h>

typedef __attribute__((ext_vector_type(16))) __bf16          v16bf;
typedef __attribute__((ext_vector_type(8)))  float           v8f;
typedef __attribute__((ext_vector_type(16))) unsigned short  v16u;
typedef __attribute__((ext_vector_type(8)))  unsigned short  u16x8;

#define H1 64
#define H2 128
#define TSTEPS 128
#define FIN 12
#define MTILES 2                       // batch tiles (of 16 rows) per wave
#define ROWS_PER_BLOCK (MTILES * 16)   // one wave per block

// ---------- small helpers ----------

__device__ __forceinline__ unsigned short f32_to_bf16(float f) {
  unsigned int u = __builtin_bit_cast(unsigned int, f);
  u += 0x7FFFu + ((u >> 16) & 1u);           // round-to-nearest-even
  return (unsigned short)(u >> 16);
}

__device__ __forceinline__ float bf16_to_f32(unsigned short s) {
  unsigned int u = ((unsigned int)s) << 16;
  return __builtin_bit_cast(float, u);
}

__device__ __forceinline__ float sigm(float x) {
  return 1.0f / (1.0f + __expf(-x));
}

__device__ __forceinline__ float tanh_fast(float x) {
  float ax = fabsf(x);
  float e  = __expf(-2.0f * ax);
  float t  = 1.0f - 2.0f * e / (1.0f + e);
  return copysignf(t, x);
}

__device__ __forceinline__ v8f splat8(float v) {
  v8f r = {v, v, v, v, v, v, v, v};
  return r;
}

__device__ __forceinline__ v8f wmma_bf16(v16bf a, v16bf b, v8f c) {
  // (neg_a, A, neg_b, B, c_mod, C, reuse_a, reuse_b)
  return __builtin_amdgcn_wmma_f32_16x16x32_bf16(false, a, false, b, (short)0, c,
                                                 false, false);
}

// B fragment, pre-swizzled in workspace: 512 halves / fragment, lane*16 each.
__device__ __forceinline__ v16bf gfrag(const unsigned short* p) {
  return __builtin_bit_cast(v16bf, *(const v16u*)p);
}

// A fragment from bf16 row-major LDS buffer h[16][H].
// 16-bit A 16x32 layout: lane<16 -> M=lane, K = 0..7 | 16..23
//                        lane>=16 -> M=lane-16, K = 8..15 | 24..31
__device__ __forceinline__ v16bf lds_afrag(const unsigned short* hbuf, int H,
                                           int kstep, int lane) {
  int m     = lane & 15;
  int kbase = (lane & 16) ? 8 : 0;
  const u16x8* p = (const u16x8*)(hbuf + m * H + kstep * 32 + kbase);
  u16x8 lo = p[0];
  u16x8 hi = p[2];   // +16 halves
  v16u v = __builtin_shufflevector(lo, hi, 0, 1, 2, 3, 4, 5, 6, 7,
                                           8, 9, 10, 11, 12, 13, 14, 15);
  return __builtin_bit_cast(v16bf, v);
}

// A fragment for x_t (K = 12 real, padded to 32 with zeros), f32 -> bf16.
__device__ __forceinline__ v16bf load_x_frag(const float* px, int lane) {
  float4 a, b;
  if (lane < 16) {                 // K = 0..7 (all real)
    a = *(const float4*)(px);
    b = *(const float4*)(px + 4);
  } else {                         // K = 8..15 -> 8..11 real, rest 0
    a = *(const float4*)(px + 8);
    b = make_float4(0.f, 0.f, 0.f, 0.f);
  }
  v16u v;
  v[0] = f32_to_bf16(a.x); v[1] = f32_to_bf16(a.y);
  v[2] = f32_to_bf16(a.z); v[3] = f32_to_bf16(a.w);
  v[4] = f32_to_bf16(b.x); v[5] = f32_to_bf16(b.y);
  v[6] = f32_to_bf16(b.z); v[7] = f32_to_bf16(b.w);
  v[8] = 0; v[9] = 0; v[10] = 0; v[11] = 0;
  v[12] = 0; v[13] = 0; v[14] = 0; v[15] = 0;
  return __builtin_bit_cast(v16bf, v);
}

// ---------- prep kernel: swizzle weights into WMMA B-fragment layout ----------
__device__ void build_frags(const float* W, int N, int Kreal,
                            unsigned short* dst, int tid, int nth) {
  int ntiles = N >> 4;
  int ksteps = (Kreal + 31) >> 5;
  int total  = ntiles * ksteps * 512;
  for (int idx = tid; idx < total; idx += nth) {
    int h     = idx & 15;
    int lane  = (idx >> 4) & 31;
    int f     = idx >> 9;
    int kstep = f % ksteps;
    int ntile = f / ksteps;
    int n = ntile * 16 + (lane & 15);
    int k = kstep * 32 + ((lane & 16) ? 8 : 0) + ((h < 8) ? h : h + 8);
    float v = (k < Kreal) ? W[n * Kreal + k] : 0.0f;
    dst[idx] = f32_to_bf16(v);
  }
}

__global__ void lstm_prep(const float* Wih0, const float* Whh0,
                          const float* bih0, const float* bhh0,
                          const float* Wih1, const float* Whh1,
                          const float* bih1, const float* bhh1,
                          const float* Wd1, const float* bd1,
                          const float* Wd2, const float* bd2,
                          unsigned short* fih0, unsigned short* fhh0,
                          unsigned short* fih1, unsigned short* fhh1,
                          float* bias0, float* bias1, float* weff, float* beff) {
  int tid = threadIdx.x, nth = blockDim.x;
  build_frags(Wih0, 4 * H1, FIN, fih0, tid, nth);
  build_frags(Whh0, 4 * H1, H1,  fhh0, tid, nth);
  build_frags(Wih1, 4 * H2, H1,  fih1, tid, nth);
  build_frags(Whh1, 4 * H2, H2,  fhh1, tid, nth);
  for (int i = tid; i < 4 * H1; i += nth) bias0[i] = bih0[i] + bhh0[i];
  for (int i = tid; i < 4 * H2; i += nth) bias1[i] = bih1[i] + bhh1[i];
  // collapse the two linear head layers: pred = h1 . weff + beff
  for (int k = tid; k < H2; k += nth) {
    float s = 0.f;
    for (int j = 0; j < H1; ++j) s += Wd2[j] * Wd1[j * H2 + k];
    weff[k] = s;
  }
  if (tid == 0) {
    float s = bd2[0];
    for (int j = 0; j < H1; ++j) s += bd1[j] * Wd2[j];
    beff[0] = s;
  }
}

// ---------- main kernel: one wave per block, MTILES batch tiles per wave ----------

__global__ __launch_bounds__(32) void lstm_main(
    const float* __restrict__ x,
    const unsigned short* __restrict__ fih0, const unsigned short* __restrict__ fhh0,
    const unsigned short* __restrict__ fih1, const unsigned short* __restrict__ fhh1,
    const float* __restrict__ bias0, const float* __restrict__ bias1,
    const float* __restrict__ weff, const float* __restrict__ beff,
    float* __restrict__ out, int B) {
  __shared__ __align__(16) unsigned short h0s[MTILES][16 * H1];
  __shared__ __align__(16) unsigned short h1s[MTILES][16 * H2];
  __shared__ __align__(16) float          c0s[MTILES][16 * H1];
  __shared__ __align__(16) float          c1s[MTILES][16 * H2];
  __shared__ float bias0s[4 * H1];
  __shared__ float bias1s[4 * H2];

  const int lane = threadIdx.x & 31;
  const int batchBase = blockIdx.x * ROWS_PER_BLOCK;
  if (batchBase >= B) return;

  // single-wave block: stage biases, init state (LDS ops in-order within wave)
  for (int i = lane; i < 4 * H1; i += 32) bias0s[i] = bias0[i];
  for (int i = lane; i < 4 * H2; i += 32) bias1s[i] = bias1[i];
#pragma unroll
  for (int mt = 0; mt < MTILES; ++mt) {
    for (int i = lane; i < 16 * H1; i += 32) { h0s[mt][i] = 0; c0s[mt][i] = 0.f; }
    for (int i = lane; i < 16 * H2; i += 32) { h1s[mt][i] = 0; c1s[mt][i] = 0.f; }
  }

  const int col    = lane & 15;
  const int rowoff = (lane >> 4) << 3;   // 0 or 8 (C/D layout rows per lane-half)
  const int m      = lane & 15;
  const float* xrow[MTILES];
#pragma unroll
  for (int mt = 0; mt < MTILES; ++mt)
    xrow[mt] = x + (size_t)(batchBase + mt * 16 + m) * TSTEPS * FIN;

#pragma clang loop unroll(disable)
  for (int t = 0; t < TSTEPS; ++t) {
    // ================= layer 0: gates0[16,256] per tile =================
    v16bf ax[MTILES];
    v16bf ah0[MTILES][2];
#pragma unroll
    for (int mt = 0; mt < MTILES; ++mt) {
      ax[mt] = load_x_frag(xrow[mt] + t * FIN, lane);
#pragma unroll
      for (int ks = 0; ks < 2; ++ks) ah0[mt][ks] = lds_afrag(h0s[mt], H1, ks, lane);
    }

#pragma clang loop unroll(disable)
    for (int jt = 0; jt < 4; ++jt) {
      v8f acc[MTILES][4];
#pragma unroll
      for (int mt = 0; mt < MTILES; ++mt)
#pragma unroll
        for (int g = 0; g < 4; ++g)            // bias in accumulator init
          acc[mt][g] = splat8(bias0s[g * H1 + jt * 16 + col]);
#pragma unroll
      for (int g = 0; g < 4; ++g) {
        v16bf b = gfrag(fih0 + (g * 4 + jt) * 512 + lane * 16);
#pragma unroll
        for (int mt = 0; mt < MTILES; ++mt)
          acc[mt][g] = wmma_bf16(ax[mt], b, acc[mt][g]);
      }
#pragma unroll
      for (int ks = 0; ks < 2; ++ks)
#pragma unroll
        for (int g = 0; g < 4; ++g) {
          v16bf b = gfrag(fhh0 + ((g * 4 + jt) * 2 + ks) * 512 + lane * 16);
#pragma unroll
          for (int mt = 0; mt < MTILES; ++mt)
            acc[mt][g] = wmma_bf16(ah0[mt][ks], b, acc[mt][g]);
        }
#pragma unroll
      for (int mt = 0; mt < MTILES; ++mt) {
        float* cpt = &c0s[mt][rowoff * H1 + jt * 16 + col];
        unsigned short* hpt = &h0s[mt][rowoff * H1 + jt * 16 + col];
#pragma unroll
        for (int r = 0; r < 8; ++r) {
          float iv = sigm(acc[mt][0][r]);
          float fv = sigm(acc[mt][1][r]);
          float gv = tanh_fast(acc[mt][2][r]);
          float ov = sigm(acc[mt][3][r]);
          float c  = fv * cpt[r * H1] + iv * gv;
          cpt[r * H1] = c;
          hpt[r * H1] = f32_to_bf16(ov * tanh_fast(c));
        }
      }
    }

    // ================= layer 1: gates1[16,512] per tile =================
    v16bf ah0n[MTILES][2];
    v16bf ah1[MTILES][4];
#pragma unroll
    for (int mt = 0; mt < MTILES; ++mt) {
#pragma unroll
      for (int ks = 0; ks < 2; ++ks) ah0n[mt][ks] = lds_afrag(h0s[mt], H1, ks, lane);
#pragma unroll
      for (int ks = 0; ks < 4; ++ks) ah1[mt][ks] = lds_afrag(h1s[mt], H2, ks, lane);
    }

#pragma clang loop unroll(disable)
    for (int jt = 0; jt < 8; ++jt) {
      v8f acc[MTILES][4];
#pragma unroll
      for (int mt = 0; mt < MTILES; ++mt)
#pragma unroll
        for (int g = 0; g < 4; ++g)
          acc[mt][g] = splat8(bias1s[g * H2 + jt * 16 + col]);
#pragma unroll
      for (int ks = 0; ks < 2; ++ks)
#pragma unroll
        for (int g = 0; g < 4; ++g) {
          v16bf b = gfrag(fih1 + ((g * 8 + jt) * 2 + ks) * 512 + lane * 16);
#pragma unroll
          for (int mt = 0; mt < MTILES; ++mt)
            acc[mt][g] = wmma_bf16(ah0n[mt][ks], b, acc[mt][g]);
        }
#pragma unroll
      for (int ks = 0; ks < 4; ++ks)
#pragma unroll
        for (int g = 0; g < 4; ++g) {
          v16bf b = gfrag(fhh1 + ((g * 8 + jt) * 4 + ks) * 512 + lane * 16);
#pragma unroll
          for (int mt = 0; mt < MTILES; ++mt)
            acc[mt][g] = wmma_bf16(ah1[mt][ks], b, acc[mt][g]);
        }
#pragma unroll
      for (int mt = 0; mt < MTILES; ++mt) {
        float* cpt = &c1s[mt][rowoff * H2 + jt * 16 + col];
        unsigned short* hpt = &h1s[mt][rowoff * H2 + jt * 16 + col];
#pragma unroll
        for (int r = 0; r < 8; ++r) {
          float iv = sigm(acc[mt][0][r]);
          float fv = sigm(acc[mt][1][r]);
          float gv = tanh_fast(acc[mt][2][r]);
          float ov = sigm(acc[mt][3][r]);
          float c  = fv * cpt[r * H2] + iv * gv;
          cpt[r * H2] = c;
          hpt[r * H2] = f32_to_bf16(ov * tanh_fast(c));
        }
      }
    }
  }

  // ================= collapsed head: out = h1 . weff + beff =================
  if (lane < 16) {
#pragma unroll
    for (int mt = 0; mt < MTILES; ++mt) {
      float acc = beff[0];
#pragma unroll 8
      for (int k = 0; k < H2; ++k)
        acc += bf16_to_f32(h1s[mt][m * H2 + k]) * weff[k];
      out[batchBase + mt * 16 + m] = acc;
    }
  }
}

// ---------- launch ----------

extern "C" void kernel_launch(void* const* d_in, const int* in_sizes, int n_in,
                              void* d_out, int out_size, void* d_ws, size_t ws_size,
                              hipStream_t stream) {
  (void)n_in; (void)out_size; (void)ws_size;
  const float* x    = (const float*)d_in[0];
  const float* Wih0 = (const float*)d_in[1];
  const float* Whh0 = (const float*)d_in[2];
  const float* bih0 = (const float*)d_in[3];
  const float* bhh0 = (const float*)d_in[4];
  const float* Wih1 = (const float*)d_in[5];
  const float* Whh1 = (const float*)d_in[6];
  const float* bih1 = (const float*)d_in[7];
  const float* bhh1 = (const float*)d_in[8];
  const float* Wd1  = (const float*)d_in[9];
  const float* bd1  = (const float*)d_in[10];
  const float* Wd2  = (const float*)d_in[11];
  const float* bd2  = (const float*)d_in[12];

  char* ws = (char*)d_ws;
  unsigned short* fih0 = (unsigned short*)(ws + 0);        // 16 frags * 1KB
  unsigned short* fhh0 = (unsigned short*)(ws + 16384);    // 32 frags
  unsigned short* fih1 = (unsigned short*)(ws + 49152);    // 64 frags
  unsigned short* fhh1 = (unsigned short*)(ws + 114688);   // 128 frags
  float* bias0 = (float*)(ws + 245760);
  float* bias1 = (float*)(ws + 246784);
  float* weff  = (float*)(ws + 248832);
  float* beff  = (float*)(ws + 249344);

  int B = in_sizes[0] / (TSTEPS * FIN);

  lstm_prep<<<1, 256, 0, stream>>>(Wih0, Whh0, bih0, bhh0, Wih1, Whh1, bih1, bhh1,
                                   Wd1, bd1, Wd2, bd2,
                                   fih0, fhh0, fih1, fhh1, bias0, bias1, weff, beff);

  int blocks = (B + ROWS_PER_BLOCK - 1) / ROWS_PER_BLOCK;
  lstm_main<<<blocks, 32, 0, stream>>>(x, fih0, fhh0, fih1, fhh1,
                                       bias0, bias1, weff, beff,
                                       (float*)d_out, B);
}
